// LocalCrossAttention_rctc_23347442221445
// MI455X (gfx1250) — compile-verified, matrix operand order checked by reading, not run on hardware
//
#include <hip/hip_runtime.h>
#include <hip/hip_bf16.h>
#include <math.h>

// ---------------------------------------------------------------------------
// MI455X (gfx1250) implementation.
//  - All GEMMs: v_wmma_f32_16x16x32_bf16 (bf16 inputs, fp32 accum).
//  - wave32; 256-thread blocks = 8 waves; 128x64 tile per block; K step 32.
//  - Each wave: 32x32 output = 4 WMMAs per K-step (A/B fragment reuse).
//  - Tile staging: 16B/thread/matrix async global->LDS copy (ASYNCcnt),
//    DOUBLE-BUFFERED: tile k+1 copies in flight while tile k is in the WMMAs;
//    s_wait_asynccnt 3 retires only the older tile (async loads complete
//    in order). Fallback: global_load_b128 + ds_store_b128.
//  - Fragment layouts per CDNA5 ISA 7.12.2 (16-bit A 16x32, B 32x16, f32 C/D).
// ---------------------------------------------------------------------------

typedef __attribute__((ext_vector_type(16))) __bf16 v16bf;
typedef __attribute__((ext_vector_type(2)))  __bf16 bf16x2;
typedef __attribute__((ext_vector_type(8)))  float  v8f;
typedef __attribute__((ext_vector_type(4)))  unsigned u32x4;
typedef __attribute__((ext_vector_type(4)))  int      v4i;

#if __has_builtin(__builtin_amdgcn_global_load_async_to_lds_b128)
#define HAS_ASYNC_LDS 1
#else
#define HAS_ASYNC_LDS 0
#endif

__device__ __forceinline__ void copy16_g2l(const __bf16* g, __bf16* l) {
#if HAS_ASYNC_LDS
  __builtin_amdgcn_global_load_async_to_lds_b128(
      (__attribute__((address_space(1))) v4i*)g,
      (__attribute__((address_space(3))) v4i*)l, 0, 0);
#else
  *(u32x4*)l = *(const u32x4*)g;
#endif
}

// Retire async copies: keep at most `deep` (3 per in-flight tile) outstanding.
__device__ __forceinline__ void copy_fence(bool keep_next_tile) {
#if HAS_ASYNC_LDS
#if __has_builtin(__builtin_amdgcn_s_wait_asynccnt)
  if (keep_next_tile) __builtin_amdgcn_s_wait_asynccnt(3);
  else                __builtin_amdgcn_s_wait_asynccnt(0);
#else
  if (keep_next_tile) asm volatile("s_wait_asynccnt 3" ::: "memory");
  else                asm volatile("s_wait_asynccnt 0" ::: "memory");
#endif
#endif
}

__device__ __forceinline__ __bf16 f2bf(float f) {
  unsigned u = __float_as_uint(f);
  u += 0x7FFFu + ((u >> 16) & 1u);          // round-to-nearest-even
  unsigned short s = (unsigned short)(u >> 16);
  __bf16 r;
  __builtin_memcpy(&r, &s, 2);
  return r;
}

// ---------------------------------------------------------------------------
// Batched WMMA GEMM:  C[b,h] = act(alpha * A[b,h] (x) B[b,h] + bias)
//   TRANSB=0:  C[M,N] = A[M,K] * B[K,N]      (B row-major K x N)
//   TRANSB=1:  C[M,N] = A[M,K] * B[N,K]^T    (B row-major N x K)
// Requires: M%128==0, N%64==0, K%32==0, lda/ldb multiples of 8, even ldc,
//           16B-aligned base pointers, head offsets multiples of 8 elements.
// ---------------------------------------------------------------------------
template<bool TRANSB, bool OUT_BF16, bool ACT_TANH>
__global__ __launch_bounds__(256)
void gemm_bf16_wmma(const __bf16* __restrict__ A, const __bf16* __restrict__ Bm,
                    void* __restrict__ Cv, int K, int lda, int ldb, int ldc,
                    long long sAb, long long sAh, long long sBb, long long sBh,
                    long long sCb, long long sCh, int nheads,
                    const float* __restrict__ bias, float alpha)
{
  constexpr int TK = 32;
  constexpr int SA = TK + 8;       // A/B-NT LDS row stride (80B, 16B aligned)
  constexpr int SB = 64 + 8;       // B-NN LDS row stride  (144B, 16B aligned)
  __shared__ __bf16 As[2][128 * SA];           // 2 x 10240 B (ping-pong)
  __shared__ __bf16 Bs[2][64 * SA];            // 2 x  5120 B (NN uses 32*SB)

  const int bI = blockIdx.z / nheads;
  const int hI = blockIdx.z % nheads;
  const __bf16* Ab = A  + (long long)bI * sAb + (long long)hI * sAh;
  const __bf16* Bb = Bm + (long long)bI * sBb + (long long)hI * sBh;

  const int m0 = blockIdx.y * 128;
  const int n0 = blockIdx.x * 64;
  const int tid  = threadIdx.x;
  const int lane = tid & 31;
  const int wave = tid >> 5;
  const int l  = lane & 15;       // row/col within 16-wide tile
  const int hi = lane >> 4;       // K/M half selector per ISA layout
  const int wm = wave >> 1;       // 0..3 -> 32-row slice
  const int wn = wave & 1;        // 0..1 -> 32-col slice

  v8f c00 = {}, c01 = {}, c10 = {}, c11 = {};

  // ---- hoisted per-thread staging addresses ----
  const int ar = tid >> 2, ac = tid & 3;              // A: 64 rows x 4 chunks
  const __bf16* gA0 = Ab + (size_t)(m0 + ar) * lda + ac * 8;
  const __bf16* gA1 = gA0 + (size_t)64 * lda;
  const int lA0 = ar * SA + ac * 8;                   // LDS element offsets
  const int lA1 = lA0 + 64 * SA;

  const __bf16* gB;
  int lBo;
  long long gBstep;
  if (TRANSB) {                                        // B: 64 n-rows x 4 chunks
    gB = Bb + (size_t)(n0 + ar) * ldb + ac * 8;
    lBo = ar * SA + ac * 8;
    gBstep = TK;
  } else {                                             // B: 32 k-rows x 8 chunks
    const int br = tid >> 3, bc = tid & 7;
    gB = Bb + (size_t)br * ldb + n0 + bc * 8;
    lBo = br * SB + bc * 8;
    gBstep = (long long)TK * ldb;
  }

  const int arow0 = (wm * 32 + l) * SA;                // A frag rows
  const int arow1 = (wm * 32 + 16 + l) * SA;
  const int brow0 = (wn * 32 + l) * SA;                // B frag rows (NT)
  const int brow1 = (wn * 32 + 16 + l) * SA;
  const int bcol0 = wn * 32 + l;                       // B frag cols (NN)
  const int bcol1 = wn * 32 + 16 + l;

  // ---- prologue: stage tile 0 into buffer 0 ----
  copy16_g2l(gA0, &As[0][lA0]);
  copy16_g2l(gA1, &As[0][lA1]);
  copy16_g2l(gB,  &Bs[0][lBo]);
  gA0 += TK; gA1 += TK; gB += gBstep;

  int cur = 0;
  for (int k0 = 0; k0 < K; k0 += TK) {
    const bool hasNext = (k0 + TK) < K;
    if (hasNext) {                                     // stage tile k+1
      copy16_g2l(gA0, &As[cur ^ 1][lA0]);
      copy16_g2l(gA1, &As[cur ^ 1][lA1]);
      copy16_g2l(gB,  &Bs[cur ^ 1][lBo]);
      gA0 += TK; gA1 += TK; gB += gBstep;
    }
    copy_fence(hasNext);                               // retire tile k copies
    __syncthreads();

    // ---- fragments (ISA 7.12.2 16-bit layouts) ----
    const __bf16* Ac = As[cur];
    const __bf16* Bc = Bs[cur];
    v16bf a0, a1, b0, b1;
#pragma unroll
    for (int e = 0; e < 8; ++e) {
      int dw = ((e & 4) ? 8 : 0) + hi * 4 + (e & 3);   // K-pair dword index
      bf16x2 pa0 = *(const bf16x2*)&Ac[arow0 + dw * 2];
      bf16x2 pa1 = *(const bf16x2*)&Ac[arow1 + dw * 2];
      a0[2 * e] = pa0[0]; a0[2 * e + 1] = pa0[1];
      a1[2 * e] = pa1[0]; a1[2 * e + 1] = pa1[1];
      if (TRANSB) {
        bf16x2 pb0 = *(const bf16x2*)&Bc[brow0 + dw * 2];
        bf16x2 pb1 = *(const bf16x2*)&Bc[brow1 + dw * 2];
        b0[2 * e] = pb0[0]; b0[2 * e + 1] = pb0[1];
        b1[2 * e] = pb1[0]; b1[2 * e + 1] = pb1[1];
      } else {
        int kb = dw * 2;
        b0[2 * e]     = Bc[kb * SB + bcol0];
        b0[2 * e + 1] = Bc[(kb + 1) * SB + bcol0];
        b1[2 * e]     = Bc[kb * SB + bcol1];
        b1[2 * e + 1] = Bc[(kb + 1) * SB + bcol1];
      }
    }
    c00 = __builtin_amdgcn_wmma_f32_16x16x32_bf16(false, a0, false, b0, (short)0, c00, false, false);
    c01 = __builtin_amdgcn_wmma_f32_16x16x32_bf16(false, a0, false, b1, (short)0, c01, false, false);
    c10 = __builtin_amdgcn_wmma_f32_16x16x32_bf16(false, a1, false, b0, (short)0, c10, false, false);
    c11 = __builtin_amdgcn_wmma_f32_16x16x32_bf16(false, a1, false, b1, (short)0, c11, false, false);
    __syncthreads();                                   // all reads of buf done
    cur ^= 1;
  }

  // ---- epilogue: D[r + 8*hi][l] per f32 C/D layout ----
  const int gn0 = n0 + wn * 32 + l;
  const int gn1 = gn0 + 16;
  const float bv0 = bias ? bias[gn0] : 0.0f;
  const float bv1 = bias ? bias[gn1] : 0.0f;
#pragma unroll
  for (int r = 0; r < 8; ++r) {
    int gm0 = m0 + wm * 32 + hi * 8 + r;   // a0 rows
    int gm1 = gm0 + 16;                    // a1 rows
    float o00 = c00[r] * alpha + bv0;
    float o01 = c01[r] * alpha + bv1;
    float o10 = c10[r] * alpha + bv0;
    float o11 = c11[r] * alpha + bv1;
    if (ACT_TANH) { o00 = tanhf(o00); o01 = tanhf(o01); o10 = tanhf(o10); o11 = tanhf(o11); }
    if (OUT_BF16) {
      __bf16* Co = (__bf16*)Cv + (long long)bI * sCb + (long long)hI * sCh;
      Co[(size_t)gm0 * ldc + gn0] = f2bf(o00);
      Co[(size_t)gm0 * ldc + gn1] = f2bf(o01);
      Co[(size_t)gm1 * ldc + gn0] = f2bf(o10);
      Co[(size_t)gm1 * ldc + gn1] = f2bf(o11);
    } else {
      float* Co = (float*)Cv + (long long)bI * sCb + (long long)hI * sCh;
      Co[(size_t)gm0 * ldc + gn0] = o00;
      Co[(size_t)gm0 * ldc + gn1] = o01;
      Co[(size_t)gm1 * ldc + gn0] = o10;
      Co[(size_t)gm1 * ldc + gn1] = o11;
    }
  }
}

// ---------------------------------------------------------------------------
// Elementwise / reduction helper kernels (fp32 math, bf16 outputs for GEMMs)
// ---------------------------------------------------------------------------

__global__ void cvt_f32_bf16(const float* __restrict__ in, __bf16* __restrict__ out, long long n) {
  long long i = (long long)blockIdx.x * 256 + threadIdx.x;
  if (i < n) out[i] = f2bf(in[i]);
}

// dec_raw[b,w,c] = mean_h kv[b, h*W + w, c]
__global__ void kv_mean_bf16(const float* __restrict__ kv, __bf16* __restrict__ dec,
                             int C, int W, int H) {
  long long idx = (long long)blockIdx.x * 256 + threadIdx.x;   // over B*W*C
  int c = (int)(idx % C);
  long long t = idx / C;
  int w = (int)(t % W);
  long long b = t / W;
  float s = 0.0f;
  for (int h = 0; h < H; ++h)
    s += kv[(((b * H + h) * W) + w) * C + c];
  dec[idx] = f2bf(s * (1.0f / (float)H));
}

// LayerNorm over last dim (biased variance, eps=1e-6), writes bf16
__global__ void ln_rows_bf16(const float* __restrict__ x, const float* __restrict__ g,
                             const float* __restrict__ b, __bf16* __restrict__ out, int C) {
  __shared__ float r1[128];
  __shared__ float r2[128];
  const long long row = blockIdx.x;
  const float* xr = x + row * C;
  const int tid = threadIdx.x;
  float s = 0.0f, s2 = 0.0f;
  for (int i = tid; i < C; i += 128) { float v = xr[i]; s += v; s2 += v * v; }
  r1[tid] = s; r2[tid] = s2;
  __syncthreads();
  for (int k = 64; k > 0; k >>= 1) {
    if (tid < k) { r1[tid] += r1[tid + k]; r2[tid] += r2[tid + k]; }
    __syncthreads();
  }
  float mean = r1[0] / (float)C;
  float var  = r2[0] / (float)C - mean * mean;
  float rstd = rsqrtf(var + 1e-6f);
  for (int i = tid; i < C; i += 128)
    out[row * C + i] = f2bf((xr[i] - mean) * rstd * g[i] + b[i]);
}

// Row softmax (fp32 in, bf16 out)
__global__ void softmax_rows_bf16(const float* __restrict__ sc, __bf16* __restrict__ out, int n) {
  __shared__ float red[256];
  const long long row = blockIdx.x;
  const float* x = sc + row * n;
  const int tid = threadIdx.x;
  float m = -3.0e38f;
  for (int i = tid; i < n; i += 256) m = fmaxf(m, x[i]);
  red[tid] = m; __syncthreads();
  for (int k = 128; k > 0; k >>= 1) { if (tid < k) red[tid] = fmaxf(red[tid], red[tid + k]); __syncthreads(); }
  m = red[0];
  __syncthreads();
  float s = 0.0f;
  for (int i = tid; i < n; i += 256) s += expf(x[i] - m);
  red[tid] = s; __syncthreads();
  for (int k = 128; k > 0; k >>= 1) { if (tid < k) red[tid] += red[tid + k]; __syncthreads(); }
  float inv = 1.0f / red[0];
  for (int i = tid; i < n; i += 256)
    out[row * n + i] = f2bf(expf(x[i] - m) * inv);
}

// Gaussian-windowed softmax: softmax_w( score * exp(-(w-pt)^2 / (2*sigma^2)) )
__global__ void gauss_softmax_rows_bf16(const float* __restrict__ sc, const float* __restrict__ pt,
                                        __bf16* __restrict__ out, int n) {
  __shared__ float red[256];
  const long long row = blockIdx.x;
  const float* x = sc + row * n;
  const float p = pt[row];
  const int tid = threadIdx.x;
  const float inv2s2 = 1.0f / 18.0f;     // sigma = 3
  float m = -3.0e38f;
  for (int i = tid; i < n; i += 256) {
    float d = (float)i - p;
    float v = x[i] * expf(-d * d * inv2s2);
    m = fmaxf(m, v);
  }
  red[tid] = m; __syncthreads();
  for (int k = 128; k > 0; k >>= 1) { if (tid < k) red[tid] = fmaxf(red[tid], red[tid + k]); __syncthreads(); }
  m = red[0];
  __syncthreads();
  float s = 0.0f;
  for (int i = tid; i < n; i += 256) {
    float d = (float)i - p;
    float v = x[i] * expf(-d * d * inv2s2);
    s += expf(v - m);
  }
  red[tid] = s; __syncthreads();
  for (int k = 128; k > 0; k >>= 1) { if (tid < k) red[tid] += red[tid + k]; __syncthreads(); }
  float inv = 1.0f / red[0];
  for (int i = tid; i < n; i += 256) {
    float d = (float)i - p;
    float v = x[i] * expf(-d * d * inv2s2);
    out[row * n + i] = f2bf(expf(v - m) * inv);
  }
}

// pt[row] = sigmoid( dot(t[row,:], vp_w) + vp_b ) * W
__global__ void head_pt(const float* __restrict__ t, const float* __restrict__ vw,
                        const float* __restrict__ vb, float* __restrict__ pt, int C, float Wf) {
  __shared__ float red[128];
  const long long row = blockIdx.x;
  const int tid = threadIdx.x;
  float s = 0.0f;
  for (int i = tid; i < C; i += 128) s += t[row * C + i] * vw[i];
  red[tid] = s; __syncthreads();
  for (int k = 64; k > 0; k >>= 1) { if (tid < k) red[tid] += red[tid + k]; __syncthreads(); }
  if (tid == 0) {
    float z = red[0] + vb[0];
    pt[row] = (1.0f / (1.0f + expf(-z))) * Wf;
  }
}

// ---------------------------------------------------------------------------
// Host-side helpers
// ---------------------------------------------------------------------------
template<bool TB, bool OB, bool TH>
static void run_gemm(hipStream_t st, const void* A, const void* B, void* C,
                     int M, int N, int K, int lda, int ldb, int ldc,
                     long long sAb, long long sAh, long long sBb, long long sBh,
                     long long sCb, long long sCh, int batches, int nheads,
                     const float* bias, float alpha)
{
  dim3 grid((unsigned)(N / 64), (unsigned)(M / 128), (unsigned)(batches * nheads));
  gemm_bf16_wmma<TB, OB, TH><<<grid, 256, 0, st>>>(
      (const __bf16*)A, (const __bf16*)B, C, K, lda, ldb, ldc,
      sAb, sAh, sBb, sBh, sCb, sCh, nheads, bias, alpha);
}

extern "C" void kernel_launch(void* const* d_in, const int* in_sizes, int n_in,
                              void* d_out, int out_size, void* d_ws, size_t ws_size,
                              hipStream_t stream)
{
  (void)in_sizes; (void)n_in; (void)out_size; (void)ws_size;
  const int Bsz = 32, S = 128, C = 384, H = 8, W = 512;
  const int BS = Bsz * S;                 // 4096
  const int BW = Bsz * W;                 // 16384
  const int C2 = 2 * C;                   // 768
  const int HD = C / 2;                   // 192 (2-head dim)
  const float scale1 = 1.0f / sqrtf((float)C);
  const float scale2 = 1.0f / sqrtf((float)HD);

  const float* q         = (const float*)d_in[0];
  const float* kv        = (const float*)d_in[1];
  const float* Wq        = (const float*)d_in[2];
  const float* Wkv       = (const float*)d_in[3];
  const float* Wproj     = (const float*)d_in[4];
  const float* bproj     = (const float*)d_in[5];
  const float* Wp_w      = (const float*)d_in[6];
  const float* Wp_b      = (const float*)d_in[7];
  const float* vp_w      = (const float*)d_in[8];
  const float* vp_b      = (const float*)d_in[9];
  const float* Wqpos     = (const float*)d_in[10];
  const float* bqpos     = (const float*)d_in[11];
  const float* Wrctc     = (const float*)d_in[12];
  const float* brctc     = (const float*)d_in[13];
  const float* ca1_Wq    = (const float*)d_in[14];
  const float* ca1_Wkv   = (const float*)d_in[15];
  const float* ca1_Wproj = (const float*)d_in[16];
  const float* ca1_bproj = (const float*)d_in[17];
  const float* ca2_Wq    = (const float*)d_in[18];
  const float* ca2_Wkv   = (const float*)d_in[19];
  const float* ca2_Wproj = (const float*)d_in[20];
  const float* ca2_bproj = (const float*)d_in[21];
  const float* g_q1  = (const float*)d_in[22];
  const float* b_q1  = (const float*)d_in[23];
  const float* g_kv1 = (const float*)d_in[24];
  const float* b_kv1 = (const float*)d_in[25];
  const float* g_q2  = (const float*)d_in[26];
  const float* b_q2  = (const float*)d_in[27];
  const float* g_kv2 = (const float*)d_in[28];
  const float* b_kv2 = (const float*)d_in[29];

  // ---- workspace bump allocator (256B aligned) ----
  size_t off = 0;
  char* base = (char*)d_ws;
  auto take = [&](size_t bytes) -> void* {
    void* p = base + off;
    off = (off + bytes + 255) & ~(size_t)255;
    return p;
  };

  __bf16* DEC_RAW_BF = (__bf16*)take((size_t)BW * C * 2);    // later reused as LN(dec)
  float*  DEC_F      = (float*) take((size_t)BW * C * 4);
  __bf16* DEC_BF     = (__bf16*)take((size_t)BW * C * 2);
  __bf16* KV1_BF     = (__bf16*)take((size_t)BW * C2 * 2);   // main k|v (fused)
  __bf16* Q_BF       = (__bf16*)take((size_t)BS * C * 2);
  __bf16* QH_BF      = (__bf16*)take((size_t)BS * C * 2);    // scaled q proj
  float*  P_F        = (float*) take((size_t)BS * C * 4);    // p0 then p1
  __bf16* LNA_BF     = (__bf16*)take((size_t)BS * C * 2);
  __bf16* LNB_BF     = (__bf16*)take((size_t)BS * C * 2);
  __bf16* QP_BF      = (__bf16*)take((size_t)BS * C * 2);    // ca q proj (reused)
  __bf16* KVC1_BF    = (__bf16*)take((size_t)BS * C2 * 2);
  __bf16* KVC2_BF    = (__bf16*)take((size_t)BW * C2 * 2);
  float*  SC_F       = (float*) take((size_t)64 * 128 * 512 * 4);  // score buffer (reused 3x)
  __bf16* ATT_BF     = (__bf16*)take((size_t)64 * 128 * 512 * 2);  // attn probs (reused 3x)
  __bf16* X_BF       = (__bf16*)take((size_t)BS * C * 2);          // attn output (reused 3x)
  __bf16* P2_BF      = (__bf16*)take((size_t)BS * C * 2);
  float*  T_F        = (float*) take((size_t)BS * C * 4);
  float*  PT_F       = (float*) take((size_t)BS * 4);
  // bf16 weight pool
  __bf16* WqB    = (__bf16*)take((size_t)C * C * 2);
  __bf16* WkvB   = (__bf16*)take((size_t)C * C2 * 2);
  __bf16* WprojB = (__bf16*)take((size_t)C * C * 2);
  __bf16* WqposB = (__bf16*)take((size_t)C * C * 2);
  __bf16* WrctcB = (__bf16*)take((size_t)C * C * 2);
  __bf16* c1WqB  = (__bf16*)take((size_t)C * C * 2);
  __bf16* c1WkvB = (__bf16*)take((size_t)C * C2 * 2);
  __bf16* c1WpB  = (__bf16*)take((size_t)C * C * 2);
  __bf16* c2WqB  = (__bf16*)take((size_t)C * C * 2);
  __bf16* c2WkvB = (__bf16*)take((size_t)C * C2 * 2);
  __bf16* c2WpB  = (__bf16*)take((size_t)C * C * 2);
  __bf16* WpwB   = (__bf16*)take((size_t)C * C * 2);

  auto cvt = [&](const float* in, __bf16* out, long long n) {
    cvt_f32_bf16<<<dim3((unsigned)((n + 255) / 256)), 256, 0, stream>>>(in, out, n);
  };

  // ---- 0. weight + q conversions ----
  cvt(Wq, WqB, (long long)C * C);
  cvt(Wkv, WkvB, (long long)C * C2);
  cvt(Wproj, WprojB, (long long)C * C);
  cvt(Wqpos, WqposB, (long long)C * C);
  cvt(Wrctc, WrctcB, (long long)C * C);
  cvt(ca1_Wq, c1WqB, (long long)C * C);
  cvt(ca1_Wkv, c1WkvB, (long long)C * C2);
  cvt(ca1_Wproj, c1WpB, (long long)C * C);
  cvt(ca2_Wq, c2WqB, (long long)C * C);
  cvt(ca2_Wkv, c2WkvB, (long long)C * C2);
  cvt(ca2_Wproj, c2WpB, (long long)C * C);
  cvt(Wp_w, WpwB, (long long)C * C);
  cvt(q, Q_BF, (long long)BS * C);

  // ---- 1. dec_raw = mean_h(kv)  (the 201 MB HBM-bound pass) ----
  kv_mean_bf16<<<(unsigned)(((long long)BW * C) / 256), 256, 0, stream>>>(kv, DEC_RAW_BF, C, W, H);

  // ---- 2. dec = dec_raw @ Wrctc + brctc ----
  run_gemm<false,false,false>(stream, DEC_RAW_BF, WrctcB, DEC_F,
      BW, C, C, C, C, C, 0,0,0,0,0,0, 1,1, brctc, 1.0f);
  cvt(DEC_F, DEC_BF, (long long)BW * C);

  // ---- 3. qh = (q @ Wq) * scale1 ;  kv1 = dec @ Wkv ;  p0 = q @ Wqpos + b ----
  run_gemm<false,true,false>(stream, Q_BF, WqB, QH_BF,
      BS, C, C, C, C, C, 0,0,0,0,0,0, 1,1, nullptr, scale1);
  run_gemm<false,true,false>(stream, DEC_BF, WkvB, KV1_BF,
      BW, C2, C, C, C2, C2, 0,0,0,0,0,0, 1,1, nullptr, 1.0f);
  run_gemm<false,false,false>(stream, Q_BF, WqposB, P_F,
      BS, C, C, C, C, C, 0,0,0,0,0,0, 1,1, bqpos, 1.0f);

  // ---- 4. cross-attn 1:  p = CA(LN(p0), LN(q)) ----
  ln_rows_bf16<<<(unsigned)BS, 128, 0, stream>>>(P_F, g_q1, b_q1, LNA_BF, C);
  ln_rows_bf16<<<(unsigned)BS, 128, 0, stream>>>(q,   g_kv1, b_kv1, LNB_BF, C);
  run_gemm<false,true,false>(stream, LNA_BF, c1WqB, QP_BF,
      BS, C, C, C, C, C, 0,0,0,0,0,0, 1,1, nullptr, 1.0f);
  run_gemm<false,true,false>(stream, LNB_BF, c1WkvB, KVC1_BF,
      BS, C2, C, C, C2, C2, 0,0,0,0,0,0, 1,1, nullptr, 1.0f);
  // scores[b,h] = (q_h * scale2) @ k_h^T  : batch=32, heads=2
  run_gemm<true,false,false>(stream, QP_BF, KVC1_BF, SC_F,
      S, S, HD, C, C2, S,
      (long long)S*C, HD, (long long)S*C2, HD, (long long)2*S*S, (long long)S*S,
      Bsz, 2, nullptr, scale2);
  softmax_rows_bf16<<<(unsigned)(Bsz * 2 * S), 256, 0, stream>>>(SC_F, ATT_BF, S);
  run_gemm<false,true,false>(stream, ATT_BF, KVC1_BF + C /* v half */, X_BF,
      S, HD, S, S, C2, C,
      (long long)2*S*S, (long long)S*S, (long long)S*C2, HD, (long long)S*C, HD,
      Bsz, 2, nullptr, 1.0f);
  run_gemm<false,false,false>(stream, X_BF, c1WpB, P_F,
      BS, C, C, C, C, C, 0,0,0,0,0,0, 1,1, ca1_bproj, 1.0f);

  // ---- 5. cross-attn 2:  p = CA(LN(p1), LN(dec)) ----
  ln_rows_bf16<<<(unsigned)BS, 128, 0, stream>>>(P_F, g_q2, b_q2, LNA_BF, C);
  ln_rows_bf16<<<(unsigned)BW, 128, 0, stream>>>(DEC_F, g_kv2, b_kv2, DEC_RAW_BF /* LN(dec) */, C);
  run_gemm<false,true,false>(stream, LNA_BF, c2WqB, QP_BF,
      BS, C, C, C, C, C, 0,0,0,0,0,0, 1,1, nullptr, 1.0f);
  run_gemm<false,true,false>(stream, DEC_RAW_BF, c2WkvB, KVC2_BF,
      BW, C2, C, C, C2, C2, 0,0,0,0,0,0, 1,1, nullptr, 1.0f);
  run_gemm<true,false,false>(stream, QP_BF, KVC2_BF, SC_F,
      S, W, HD, C, C2, W,
      (long long)S*C, HD, (long long)W*C2, HD, (long long)2*S*W, (long long)S*W,
      Bsz, 2, nullptr, scale2);
  softmax_rows_bf16<<<(unsigned)(Bsz * 2 * S), 256, 0, stream>>>(SC_F, ATT_BF, W);
  run_gemm<false,true,false>(stream, ATT_BF, KVC2_BF + C, X_BF,
      S, HD, W, W, C2, C,
      (long long)2*S*W, (long long)S*W, (long long)W*C2, HD, (long long)S*C, HD,
      Bsz, 2, nullptr, 1.0f);
  run_gemm<false,true,false>(stream, X_BF, c2WpB, P2_BF,
      BS, C, C, C, C, C, 0,0,0,0,0,0, 1,1, ca2_bproj, 1.0f);

  // ---- 6. p_t = sigmoid(tanh(p2 @ Wp_w + Wp_b) @ vp_w + vp_b) * W ----
  run_gemm<false,false,true>(stream, P2_BF, WpwB, T_F,
      BS, C, C, C, C, C, 0,0,0,0,0,0, 1,1, Wp_b, 1.0f);
  head_pt<<<(unsigned)BS, 128, 0, stream>>>(T_F, vp_w, vp_b, PT_F, C, (float)W);

  // ---- 7. main attention with gaussian window ----
  run_gemm<true,false,false>(stream, QH_BF, KV1_BF, SC_F,
      S, W, C, C, C2, W,
      (long long)S*C, 0, (long long)W*C2, 0, (long long)S*W, 0,
      Bsz, 1, nullptr, 1.0f);
  gauss_softmax_rows_bf16<<<(unsigned)BS, 256, 0, stream>>>(SC_F, PT_F, ATT_BF, W);
  run_gemm<false,true,false>(stream, ATT_BF, KV1_BF + C, X_BF,
      S, C, W, W, C2, C,
      (long long)S*W, 0, (long long)W*C2, 0, (long long)S*C, 0,
      Bsz, 1, nullptr, 1.0f);

  // ---- 8. out = x @ Wproj + bproj ----
  run_gemm<false,false,false>(stream, X_BF, WprojB, (float*)d_out,
      BS, C, C, C, C, C, 0,0,0,0,0,0, 1,1, bproj, 1.0f);
}
